// SteeredConvTranspose2d_78769700209280
// MI455X (gfx1250) — matrix operand order repaired and checked by reading, not compiled, not run on hardware
//
#include <hip/hip_runtime.h>
#include <math.h>

typedef __attribute__((ext_vector_type(8)))  float        v8f;
typedef __attribute__((ext_vector_type(8)))  __bf16       v8bf;
typedef __attribute__((ext_vector_type(16))) __bf16       v16bf;
typedef __attribute__((ext_vector_type(4)))  unsigned int v4u;
typedef __attribute__((ext_vector_type(8)))  int          v8i;
typedef __attribute__((ext_vector_type(4)))  int          v4i;

static __device__ __forceinline__ v16bf bfcat(v8bf a, v8bf b) {
  return __builtin_shufflevector(a, b, 0, 1, 2, 3, 4, 5, 6, 7, 8, 9, 10, 11,
                                 12, 13, 14, 15);
}

// ---------------------------------------------------------------------------
// Workspace layout (bytes):
//   uv   : f32 [8][128][2][2][64][64] parity-major      = 67,108,864
//   KA2h/KA2l : bf16 [25 tap][320 M][64 k]  hi/lo       =  2,048,000
//   KA1h/KA1l : bf16 [49 tap][128 M][64 k]  hi/lo       =  1,605,632
//   xph/xpl   : bf16 [8][68][68][64] NHWC zero-padded   =  9,469,952
// ---------------------------------------------------------------------------

// ---------------------------------------------------------------------------
// TDM: async-load a 3D tile (dim0 = 64 ch contiguous, dim1 = cols, dim2 =
// rows) from the padded NHWC bf16 buffer into LDS. pad_enable inserts 16 B
// after every 64 ch (32 DWORDs) -> LDS pixel stride 72 bf16 (144 B), matching
// the GEMM's bank-spread addressing. Tracked by TENSORcnt.
// ---------------------------------------------------------------------------
static __device__ __forceinline__ void tdm_load_tile(unsigned lds_off,
                                                     const __bf16* gptr,
                                                     unsigned tile1,
                                                     unsigned tile2) {
  unsigned long long ga = (unsigned long long)gptr;
  v4u g0;
  g0[0] = 1u;                                      // count=1, user descriptor
  g0[1] = lds_off;                                 // LDS byte address
  g0[2] = (unsigned)(ga & 0xffffffffu);            // global_addr[31:0]
  g0[3] = (unsigned)((ga >> 32) & 0x01ffffffu) | (2u << 30);  // addr hi | type=2
  v8i g1;
  g1[0] = (int)((1u << 16)    // data_size = 2 bytes
              | (1u << 20)    // pad_enable
              | (4u << 22)    // pad_interval code 4 = 32 DWORDs (one pixel)
              | (3u << 25));  // pad_amount  code 3 = 4 DWORDs (16 B)
  g1[1] = (int)(64u << 16);                        // tensor_dim0 = 64 (lo16)
  g1[2] = (int)(tile1 << 16);                      // dim0 hi=0 | tensor_dim1 lo16
  g1[3] = (int)(64u << 16);                        // tensor_dim1 hi=0 | tile_dim0=64
  g1[4] = (int)(tile1 | (tile2 << 16));            // tile_dim1 | tile_dim2
  g1[5] = (int)64;                                 // tensor_dim0_stride = 64
  g1[6] = (int)((4352u & 0xffffu) << 16);          // s0 hi=0 | tensor_dim1_stride lo16
  g1[7] = (int)(4352u >> 16);                      // tensor_dim1_stride[47:16]
  v4i g2;
  g2[0] = (int)tile2;                              // tensor_dim2
  g2[1] = 0;
  g2[2] = 0;
  g2[3] = 0;
  v4i g3 = {0, 0, 0, 0};
#if defined(__clang_major__) && (__clang_major__ >= 23)
  v8i g4 = {0, 0, 0, 0, 0, 0, 0, 0};
  __builtin_amdgcn_tensor_load_to_lds(g0, g1, g2, g3, g4, 0);
#else
  __builtin_amdgcn_tensor_load_to_lds(g0, g1, g2, g3, 0);
#endif
}

// ----------------------------- basis helpers ------------------------------
__device__ __forceinline__ float prof5(float rad, int h, int w) {
  float ys = (float)(h - 2), xs = (float)(w - 2);
  float rr = sqrtf(ys * ys + xs * xs);
  float d = rr - rad;
  return expf(-2.0f * d * d);            // SIG=0.5 -> 1/(2*sig^2) = 2
}
__device__ __forceinline__ float norm5(float rad) {
  float s = 0.f;
  for (int h = 0; h < 5; ++h)
    for (int w = 0; w < 5; ++w) s += prof5(rad, h, w);
  return s;
}
__device__ __forceinline__ float prof7(float rad, int h, int w) {
  float ys = (float)(h - 3), xs = (float)(w - 3);
  float rr = sqrtf(ys * ys + xs * xs);
  float d = rr - rad;
  return expf(-2.0f * d * d);
}
__device__ __forceinline__ float norm7(float rad) {
  float s = 0.f;
  for (int h = 0; h < 7; ++h)
    for (int w = 0; w < 7; ++w) s += prof7(rad, h, w);
  return s;
}

// ------------- pre-split x into zero-padded NHWC bf16 hi/lo ---------------
__global__ void split_x_kernel(const float* __restrict__ x,
                               __bf16* __restrict__ xph,
                               __bf16* __restrict__ xpl) {
  int idx = blockIdx.x * blockDim.x + threadIdx.x;
  if (idx >= 8 * 68 * 68 * 64) return;
  int ch = idx & 63;
  int t = idx >> 6;
  int xc = t % 68;
  t /= 68;
  int yc = t % 68;
  int b = t / 68;
  int y = yc - 2, xx = xc - 2;
  float v = 0.f;
  if ((unsigned)y < 64u && (unsigned)xx < 64u)
    v = x[((b * 64 + ch) * 64 + y) * 64 + xx];
  __bf16 hi = (__bf16)v;
  __bf16 lo = (__bf16)(v - (float)hi);
  xph[idx] = hi;
  xpl[idx] = lo;
}

// ---------------- compose Kall (320,64,5,5) -> bf16 hi/lo [t][M][k] -------
__global__ void build_KA2_kernel(const float* __restrict__ w0,
                                 const float* __restrict__ wc,
                                 const float* __restrict__ wsn,
                                 __bf16* __restrict__ KAh,
                                 __bf16* __restrict__ KAl) {
  int idx = blockIdx.x * blockDim.x + threadIdx.x;
  if (idx >= 25 * 320 * 64) return;
  int t = idx / (320 * 64);
  int rem = idx % (320 * 64);
  int m = rem / 64;           // output row: g*64 + o
  int k = rem % 64;           // input channel
  int h = t / 5, w = t % 5;
  int g = m / 64, o = m % 64;
  float val = 0.f;
  if (g == 0) {
    for (int r = 0; r < 3; ++r) {
      float rad = (float)r;
      val += w0[(o * 64 + k) * 3 + r] * (prof5(rad, h, w) / norm5(rad));
    }
  } else {
    int kk = (g - 1) >> 1;          // harmonic index (k = kk+1)
    int issin = (g - 1) & 1;        // g=1:cos g=2:sin g=3:cos g=4:sin
    float phi = atan2f((float)(h - 2), (float)(w - 2));
    float tr = issin ? sinf((float)(kk + 1) * phi) : cosf((float)(kk + 1) * phi);
    const float* wptr = issin ? wsn : wc;
    for (int r = 0; r < 2; ++r) {
      float rad = (float)(r + 1);
      val += wptr[((kk * 64 + o) * 64 + k) * 2 + r] *
             (prof5(rad, h, w) / norm5(rad)) * tr;
    }
  }
  __bf16 hi = (__bf16)val;
  __bf16 lo = (__bf16)(val - (float)hi);
  KAh[idx] = hi;
  KAl[idx] = lo;
}

// ---------------- compose Ka (128,64,7,7) -> bf16 hi/lo [t][M][k] ---------
__global__ void build_KA1_kernel(const float* __restrict__ wa,
                                 __bf16* __restrict__ KAh,
                                 __bf16* __restrict__ KAl) {
  int idx = blockIdx.x * blockDim.x + threadIdx.x;
  if (idx >= 49 * 128 * 64) return;
  int t = idx / (128 * 64);
  int rem = idx % (128 * 64);
  int m = rem / 64;           // 0..63 cos (u), 64..127 sin (v)
  int k = rem % 64;
  int h = t / 7, w = t % 7;
  int o = m & 63;
  int issin = m >> 6;
  float phi = atan2f((float)(h - 3), (float)(w - 3));
  float tr = issin ? sinf(phi) : cosf(phi);
  float val = 0.f;
  for (int r = 0; r < 3; ++r) {
    float rad = (float)(r + 1);
    val += wa[(o * 64 + k) * 3 + r] * (prof7(rad, h, w) / norm7(rad)) * tr;
  }
  __bf16 hi = (__bf16)val;
  __bf16 lo = (__bf16)(val - (float)hi);
  KAh[idx] = hi;
  KAl[idx] = lo;
}

// ---------------------------------------------------------------------------
// conv_uv: transposed conv 7x7 stride 2 pad (4,3), M = 128 (u,v channels).
// Phase (a,c): p = 2p'+a, q = 2q'+c. Patch [5r][68c][72 ch-pad] bf16 hi/lo
// staged by TDM. Split-precision GEMM: acc += Ah*Bh + Ah*Bl + Al*Bh.
// ---------------------------------------------------------------------------
__global__ __launch_bounds__(256) void conv_uv_kernel(
    const __bf16* __restrict__ xph, const __bf16* __restrict__ xpl,
    const __bf16* __restrict__ KAh, const __bf16* __restrict__ KAl,
    float* __restrict__ uv) {
  extern __shared__ float smemf[];
  __bf16* xsh = (__bf16*)smemf;            // [340 pix][72]
  __bf16* xsl = xsh + 340 * 72;
  const int pp = blockIdx.x;
  const int a = blockIdx.y >> 1;
  const int c = blockIdx.y & 1;
  const int b = blockIdx.z;
  const int tid = threadIdx.x;
  const int wv = tid >> 5;

  // TDM-stage rows y = p'-2..p'+2 (padded rows pp..pp+4), all 68 cols.
  if (wv == 0) {
    size_t goff = ((size_t)(b * 68 + pp) * 68) * 64;
    tdm_load_tile(0u, xph + goff, 68u, 5u);
    tdm_load_tile(48960u, xpl + goff, 68u, 5u);
    __builtin_amdgcn_s_wait_tensorcnt(0);
  }
  __syncthreads();

  const int lane = tid & 31;
  const int s = lane >> 4;
  const int l = lane & 15;
  const int ny = a ? 3 : 4;
  const int nx = c ? 3 : 4;
  const int m0 = wv * 16;

  for (int nt = 0; nt < 4; ++nt) {
    v8f acc = {0.f, 0.f, 0.f, 0.f, 0.f, 0.f, 0.f, 0.f};
    for (int jy = 0; jy < ny; ++jy) {
      int r = a ? (jy + 1) : jy;
      int h = a ? (2 * jy + 1) : (2 * jy);
      for (int jx = 0; jx < nx; ++jx) {
        int cb = c ? (jx + 1) : jx;
        int w_ = c ? (2 * jx + 1) : (2 * jx);
        int tap = h * 7 + w_;
        const __bf16* Ab = KAh + ((size_t)tap * 128 + m0 + l) * 64 + 8 * s;
        const __bf16* Alb = KAl + ((size_t)tap * 128 + m0 + l) * 64 + 8 * s;
        const __bf16* Bb = xsh + (size_t)(r * 68 + cb + nt * 16 + l) * 72 + 8 * s;
        const __bf16* Blb = xsl + (size_t)(r * 68 + cb + nt * 16 + l) * 72 + 8 * s;
#pragma unroll
        for (int k0 = 0; k0 < 64; k0 += 32) {
          v16bf Ah = bfcat(*(const v8bf*)(Ab + k0), *(const v8bf*)(Ab + k0 + 16));
          v16bf Al = bfcat(*(const v8bf*)(Alb + k0), *(const v8bf*)(Alb + k0 + 16));
          v16bf Bh = bfcat(*(const v8bf*)(Bb + k0), *(const v8bf*)(Bb + k0 + 16));
          v16bf Bl = bfcat(*(const v8bf*)(Blb + k0), *(const v8bf*)(Blb + k0 + 16));
          acc = __builtin_amdgcn_wmma_f32_16x16x32_bf16(false, Ah, false, Bh,
                                                        (short)0, acc, false, false);
          acc = __builtin_amdgcn_wmma_f32_16x16x32_bf16(false, Ah, false, Bl,
                                                        (short)0, acc, false, false);
          acc = __builtin_amdgcn_wmma_f32_16x16x32_bf16(false, Al, false, Bh,
                                                        (short)0, acc, false, false);
        }
      }
    }
    int qp = nt * 16 + l;
#pragma unroll
    for (int v = 0; v < 8; ++v) {
      int ch = m0 + v + 8 * s;
      uv[((((size_t)b * 128 + ch) * 2 + a) * 2 + c) * 4096 + pp * 64 + qp] =
          acc[v];
    }
  }
}

// ---------------------------------------------------------------------------
// conv_main: transposed conv 5x5 stride 2 pad (3,2), M = 320 = 5 groups x 64,
// fused with the steering combine + bias.
// LDS: xsh/xsl [3r][66c][72] bf16 (57,024 B, TDM-staged) + ys[320][64] f32.
// ---------------------------------------------------------------------------
__global__ __launch_bounds__(256) void conv_main_kernel(
    const __bf16* __restrict__ xph, const __bf16* __restrict__ xpl,
    const __bf16* __restrict__ KAh, const __bf16* __restrict__ KAl,
    const float* __restrict__ uv, const float* __restrict__ bias,
    float* __restrict__ out) {
  extern __shared__ float smemf[];
  __bf16* xsh = (__bf16*)smemf;            // [198 pix][72]
  __bf16* xsl = xsh + 198 * 72;
  float* ys = smemf + 14256;               // after 2*198*72 bf16 = 14256 floats
  const int pp = blockIdx.x;
  const int a = blockIdx.y >> 1;
  const int c = blockIdx.y & 1;
  const int b = blockIdx.z;
  const int tid = threadIdx.x;
  const int wv = tid >> 5;

  // TDM-stage rows y = p'-1..p'+1 (padded rows pp+1..pp+3), cols x' = -1..64
  // (padded cols 1..66).
  if (wv == 0) {
    size_t goff = ((size_t)(b * 68 + pp + 1) * 68 + 1) * 64;
    tdm_load_tile(0u, xph + goff, 66u, 3u);
    tdm_load_tile(28512u, xpl + goff, 66u, 3u);
    __builtin_amdgcn_s_wait_tensorcnt(0);
  }
  __syncthreads();

  const int lane = tid & 31;
  const int s = lane >> 4;
  const int l = lane & 15;
  const int ny = a ? 3 : 2;   // a=0: h in {1,3}; a=1: h in {0,2,4}
  const int nx = c ? 3 : 2;

  for (int j = 0; j < 10; ++j) {
    int t = wv * 10 + j;
    int mt = t >> 2;          // 0..19
    int nt = t & 3;           // 0..3
    v8f acc = {0.f, 0.f, 0.f, 0.f, 0.f, 0.f, 0.f, 0.f};
    for (int jy = 0; jy < ny; ++jy) {
      int r = jy;                                  // dy = jy-1 -> r = dy+1
      int h = a ? (2 * jy) : (2 * jy + 1);
      for (int jx = 0; jx < nx; ++jx) {
        int cb = jx;                               // dx+1
        int w_ = c ? (2 * jx) : (2 * jx + 1);
        int tap = h * 5 + w_;
        const __bf16* Ab = KAh + ((size_t)tap * 320 + mt * 16 + l) * 64 + 8 * s;
        const __bf16* Alb = KAl + ((size_t)tap * 320 + mt * 16 + l) * 64 + 8 * s;
        const __bf16* Bb = xsh + (size_t)(r * 66 + cb + nt * 16 + l) * 72 + 8 * s;
        const __bf16* Blb = xsl + (size_t)(r * 66 + cb + nt * 16 + l) * 72 + 8 * s;
#pragma unroll
        for (int k0 = 0; k0 < 64; k0 += 32) {
          v16bf Ah = bfcat(*(const v8bf*)(Ab + k0), *(const v8bf*)(Ab + k0 + 16));
          v16bf Al = bfcat(*(const v8bf*)(Alb + k0), *(const v8bf*)(Alb + k0 + 16));
          v16bf Bh = bfcat(*(const v8bf*)(Bb + k0), *(const v8bf*)(Bb + k0 + 16));
          v16bf Bl = bfcat(*(const v8bf*)(Blb + k0), *(const v8bf*)(Blb + k0 + 16));
          acc = __builtin_amdgcn_wmma_f32_16x16x32_bf16(false, Ah, false, Bh,
                                                        (short)0, acc, false, false);
          acc = __builtin_amdgcn_wmma_f32_16x16x32_bf16(false, Ah, false, Bl,
                                                        (short)0, acc, false, false);
          acc = __builtin_amdgcn_wmma_f32_16x16x32_bf16(false, Al, false, Bh,
                                                        (short)0, acc, false, false);
        }
      }
    }
#pragma unroll
    for (int v = 0; v < 8; ++v)
      ys[(mt * 16 + v + 8 * s) * 64 + nt * 16 + l] = acc[v];
  }
  __syncthreads();

  // Fused steering combine: out = y0 + tanh(rho)*(c1*y1+s1*y2+c2*y3+s2*y4)+bias
  for (int i = tid; i < 64 * 64; i += 256) {
    int o = i >> 6;
    int qp = i & 63;
    float y0 = ys[(0 * 64 + o) * 64 + qp];
    float y1 = ys[(1 * 64 + o) * 64 + qp];
    float y2 = ys[(2 * 64 + o) * 64 + qp];
    float y3 = ys[(3 * 64 + o) * 64 + qp];
    float y4 = ys[(4 * 64 + o) * 64 + qp];
    float u = uv[((((size_t)b * 128 + o) * 2 + a) * 2 + c) * 4096 + pp * 64 + qp];
    float vv = uv[((((size_t)b * 128 + 64 + o) * 2 + a) * 2 + c) * 4096 + pp * 64 + qp];
    float rho_raw = sqrtf(u * u + vv * vv + 1e-8f);
    float inv = 1.0f / rho_raw;
    float un = u * inv, vn = vv * inv;
    float rho = tanhf(rho_raw);
    float c2v = un * un - vn * vn;
    float s2v = 2.0f * un * vn;
    float val = y0 + rho * (un * y1 + vn * y2 + c2v * y3 + s2v * y4) + bias[o];
    int p = 2 * pp + a;
    int q = 2 * qp + c;
    out[((size_t)(b * 64 + o) * 128 + p) * 128 + q] = val;
  }
}

// ---------------------------------------------------------------------------
extern "C" void kernel_launch(void* const* d_in, const int* in_sizes, int n_in,
                              void* d_out, int out_size, void* d_ws,
                              size_t ws_size, hipStream_t stream) {
  const float* x    = (const float*)d_in[0];
  const float* w0   = (const float*)d_in[1];
  const float* wc   = (const float*)d_in[2];
  const float* wsn  = (const float*)d_in[3];   // 'ws' weights from reference
  const float* wa   = (const float*)d_in[4];
  const float* bias = (const float*)d_in[5];
  float* out = (float*)d_out;

  char* wsb = (char*)d_ws;
  float* uv    = (float*)wsb;                        // 67,108,864 B
  __bf16* KA2h = (__bf16*)(wsb + 67108864);          // 512,000 bf16
  __bf16* KA2l = KA2h + 512000;
  __bf16* KA1h = KA2l + 512000;                      // 401,408 bf16
  __bf16* KA1l = KA1h + 401408;
  __bf16* xph  = KA1l + 401408;                      // 2,367,488 bf16
  __bf16* xpl  = xph + 8 * 68 * 68 * 64;

  const int lds_uv = 340 * 72 * 2 * 2;                       // 97,920 B
  const int lds_main = 198 * 72 * 2 * 2 + 320 * 64 * 4;      // 138,944 B
  (void)hipFuncSetAttribute((const void*)conv_uv_kernel,
                            hipFuncAttributeMaxDynamicSharedMemorySize, lds_uv);
  (void)hipFuncSetAttribute((const void*)conv_main_kernel,
                            hipFuncAttributeMaxDynamicSharedMemorySize, lds_main);

  split_x_kernel<<<9248, 256, 0, stream>>>(x, xph, xpl);
  build_KA2_kernel<<<2000, 256, 0, stream>>>(w0, wc, wsn, KA2h, KA2l);
  build_KA1_kernel<<<1568, 256, 0, stream>>>(wa, KA1h, KA1l);

  dim3 grid(64, 4, 8);   // (p' row, parity a*2+c, batch)
  conv_uv_kernel<<<grid, 256, lds_uv, stream>>>(xph, xpl, KA1h, KA1l, uv);
  conv_main_kernel<<<grid, 256, lds_main, stream>>>(xph, xpl, KA2h, KA2l, uv,
                                                    bias, out);
}
